// DiT_NN_89197880803906
// MI455X (gfx1250) — matrix operand order repaired and checked by reading, not compiled, not run on hardware
//
#include <hip/hip_runtime.h>
#include <hip/hip_bf16.h>

// ---------------- dims ----------------
#define Bz    8
#define Cc    2
#define Hh    256
#define Ww    64
#define HP    64
#define WP    16
#define Ntok  1024          // HP*WP
#define HID   1024
#define NH    16
#define HD    64
#define DEPTH 8
#define FFN   2730
#define FFN2  5460          // 2*FFN
#define FFNP  2752          // FFN padded to multiple of 32
#define TDIM  256
#define OUTC  32            // PH*PW*C
#define Mrows (Bz*Ntok)     // 8192
#define EPSV  1e-6f

typedef __attribute__((ext_vector_type(16))) _Float16 v16h;
typedef __attribute__((ext_vector_type(8)))  _Float16 v8h;
typedef __attribute__((ext_vector_type(8)))  float    v8f;

// ================= WMMA GEMM =================
// C[M,Nout] = A[M,K](f16,rowmajor) x W[Nout,K](f16,rowmajor)^T + bias
// optional epilogue: out = resid + gate[(m>>10)*gateStride + n] * val
// block = 128 threads (4 waves); each wave: 16 rows x 64 cols; K%32==0.
__global__ __launch_bounds__(128)
void dit_gemm_wmma(const _Float16* __restrict__ A,
                   const _Float16* __restrict__ Wh,
                   const float* __restrict__ bias,
                   const float* __restrict__ resid,
                   const float* __restrict__ gate, int gateStride,
                   float* __restrict__ Cout,
                   int M, int Nout, int K)
{
    const int wave  = threadIdx.x >> 5;
    const int lane  = threadIdx.x & 31;
    const int lhalf = lane >> 4;        // 0 | 1
    const int l16   = lane & 15;
    const int m0 = blockIdx.y * 64 + wave * 16;
    const int n0 = blockIdx.x * 64;
    if (m0 >= M) return;

    // A fragment rows: lane covers row (m0+l16); K-offset 0 or 16 (contiguous, matches B)
    const _Float16* aRow = A + (size_t)(m0 + l16) * K + (lhalf ? 16 : 0);

    const _Float16* bRow[4];
    int  ncol[4];
    bool nval[4];
#pragma unroll
    for (int t = 0; t < 4; ++t) {
        int n   = n0 + t * 16 + l16;
        ncol[t] = n;
        nval[t] = (n < Nout);
        int ns  = nval[t] ? n : 0;
        bRow[t] = Wh + (size_t)ns * K + (lhalf ? 16 : 0);
    }

    v8f acc[4] = {};
    for (int ks = 0; ks < K; ks += 32) {
        v8h lo = *(const v8h*)(aRow + ks);
        v8h hi = *(const v8h*)(aRow + ks + 8);
        v16h a = __builtin_shufflevector(lo, hi, 0,1,2,3,4,5,6,7,8,9,10,11,12,13,14,15);
#pragma unroll
        for (int t = 0; t < 4; ++t) {
            v16h b = *(const v16h*)(bRow[t] + ks);
            acc[t] = __builtin_amdgcn_wmma_f32_16x16x32_f16(
                         false, a, false, b, (short)0, acc[t], false, false);
        }
    }

#pragma unroll
    for (int t = 0; t < 4; ++t) {
        if (!nval[t]) continue;
        const int n  = ncol[t];
        const float bi = bias ? bias[n] : 0.f;
#pragma unroll
        for (int r = 0; r < 8; ++r) {
            int m = m0 + (lhalf ? 8 : 0) + r;   // C/D layout: lanes 16-31 -> M+8
            float v = acc[t][r] + bi;
            size_t idx = (size_t)m * Nout + n;
            if (gate) v = resid[idx] + gate[(size_t)(m >> 10) * gateStride + n] * v;
            Cout[idx] = v;
        }
    }
}

// ============ timestep embedding + MLP + SiLU -> cs[8,1024] ============
__global__ __launch_bounds__(256)
void dit_tstep(const float* __restrict__ t, const float* __restrict__ tW,
               const float* __restrict__ tB, float* __restrict__ cs)
{
    int idx = blockIdx.x * blockDim.x + threadIdx.x;
    if (idx >= Bz * HID) return;
    int b = idx >> 10, o = idx & 1023;
    float tv = t[b];
    const float* w = tW + (size_t)o * TDIM;
    float s = tB[o];
    for (int j = 0; j < 128; ++j) {
        float fr = __expf(-9.2103403719761836f * (float)j / 128.f);
        float a  = tv * fr;
        s += __cosf(a) * w[j] + __sinf(a) * w[128 + j];
    }
    cs[idx] = s / (1.f + __expf(-s));
}

// ============ patchify + pos embed -> h[8192,1024] ============
__global__ __launch_bounds__(256)
void dit_patch(const float* __restrict__ x, const float* __restrict__ pw,
               const float* __restrict__ pb, float* __restrict__ h)
{
    int idx = blockIdx.x * blockDim.x + threadIdx.x;
    if (idx >= Mrows * HID) return;
    int o = idx & 1023, bn = idx >> 10, n = bn & 1023, b = bn >> 10;
    int hp = n >> 4, wp = n & 15;
    float s = pb[o];
#pragma unroll
    for (int c2 = 0; c2 < 2; ++c2)
#pragma unroll
        for (int p = 0; p < 4; ++p)
#pragma unroll
            for (int q = 0; q < 4; ++q)
                s += x[(((size_t)(b * 2 + c2) * Hh + hp * 4 + p) * Ww) + wp * 4 + q]
                   * pw[((o * 2 + c2) * 4 + p) * 4 + q];
    float pos;
    if (o < 256)      pos = __sinf(wp * __powf(10000.f, -(float)o        / 256.f));
    else if (o < 512) pos = __cosf(wp * __powf(10000.f, -(float)(o - 256) / 256.f));
    else if (o < 768) pos = __sinf(hp * __powf(10000.f, -(float)(o - 512) / 256.f));
    else              pos = __cosf(hp * __powf(10000.f, -(float)(o - 768) / 256.f));
    h[idx] = s + pos;
}

// ============ small dense: out[8,Nout] = cs @ W^T + b ============
__global__ __launch_bounds__(256)
void dit_ada(const float* __restrict__ cs, const float* __restrict__ W,
             const float* __restrict__ bvec, float* __restrict__ out, int Nout)
{
    int idx = blockIdx.x * blockDim.x + threadIdx.x;
    if (idx >= Bz * Nout) return;
    int b = idx / Nout, j = idx - b * Nout;
    const float* c = cs + (size_t)b * HID;
    const float* w = W + (size_t)j * HID;
    float s = bvec[j];
    for (int k = 0; k < HID; ++k) s += c[k] * w[k];
    out[idx] = s;
}

// ============ RMSNorm + modulate -> fp16 A matrix ============
__global__ __launch_bounds__(256)
void dit_rmsmod(const float* __restrict__ X, const float* __restrict__ w,
                const float* __restrict__ shift, const float* __restrict__ scale,
                int sstride, _Float16* __restrict__ out)
{
    int m = blockIdx.x;            // token row
    int b = m >> 10;
    __shared__ float red[256];
    float s = 0.f;
    for (int j = threadIdx.x; j < HID; j += 256) {
        float v = X[(size_t)m * HID + j];
        s += v * v;
    }
    red[threadIdx.x] = s;
    __syncthreads();
    for (int off = 128; off > 0; off >>= 1) {
        if (threadIdx.x < off) red[threadIdx.x] += red[threadIdx.x + off];
        __syncthreads();
    }
    float rs = rsqrtf(red[0] / (float)HID + EPSV);
    for (int j = threadIdx.x; j < HID; j += 256) {
        float v = X[(size_t)m * HID + j] * rs * w[j];
        v = v * (1.f + scale[(size_t)b * sstride + j]) + shift[(size_t)b * sstride + j];
        out[(size_t)m * HID + j] = (_Float16)v;
    }
}

// ============ qkv post: split + q/k RMSNorm + RoPE -> Q,K,V [B*NH,N,HD] ============
__global__ __launch_bounds__(256)
void dit_qkvpost(const float* __restrict__ Y, const float* __restrict__ qnw,
                 const float* __restrict__ knw,
                 float* __restrict__ Q, float* __restrict__ Kb, float* __restrict__ V)
{
    int idx = blockIdx.x * blockDim.x + threadIdx.x;
    if (idx >= Bz * Ntok * NH) return;
    int h = idx & 15, bn = idx >> 4, n = bn & 1023, b = bn >> 10;
    const float* q = Y + (size_t)bn * 3072 + h * HD;
    const float* k = q + HID;
    const float* v = q + 2 * HID;
    float qs = 0.f, ks = 0.f;
    for (int e = 0; e < HD; ++e) { qs += q[e] * q[e]; ks += k[e] * k[e]; }
    float qr = rsqrtf(qs / (float)HD + EPSV);
    float kr = rsqrtf(ks / (float)HD + EPSV);
    int hp = n >> 4, wp = n & 15;
    size_t ob = ((size_t)(b * NH + h) * Ntok + n) * HD;
    for (int i = 0; i < 32; ++i) {
        float ang = (i < 16) ? hp * __powf(10000.f, -(float)i / 16.f)
                             : wp * __powf(10000.f, -(float)(i - 16) / 16.f);
        float cv = __cosf(ang), sv = __sinf(ang);
        float q0 = q[2*i]   * qr * qnw[2*i];
        float q1 = q[2*i+1] * qr * qnw[2*i+1];
        float k0 = k[2*i]   * kr * knw[2*i];
        float k1 = k[2*i+1] * kr * knw[2*i+1];
        Q[ob + 2*i]   = q0 * cv - q1 * sv;
        Q[ob + 2*i+1] = q1 * cv + q0 * sv;
        Kb[ob + 2*i]   = k0 * cv - k1 * sv;
        Kb[ob + 2*i+1] = k1 * cv + k0 * sv;
    }
    for (int e = 0; e < HD; ++e) V[ob + e] = v[e];
}

// ============ attention (streaming softmax), out fp16 [B,N,HID] ============
__global__ __launch_bounds__(128)
void dit_attn(const float* __restrict__ Q, const float* __restrict__ Kb,
              const float* __restrict__ V, _Float16* __restrict__ O)
{
    int bh = blockIdx.y;                 // b*NH + h
    int b = bh >> 4, h = bh & 15;
    int m = blockIdx.x * 128 + threadIdx.x;
    const float* qp = Q + ((size_t)bh * Ntok + m) * HD;
    float q[HD];
#pragma unroll
    for (int e = 0; e < HD; ++e) q[e] = qp[e];
    __shared__ float Ks[64][HD];
    __shared__ float Vs[64][HD];
    float acc[HD];
#pragma unroll
    for (int e = 0; e < HD; ++e) acc[e] = 0.f;
    float mx = -3.0e38f, l = 0.f;
    const float scale = 0.125f;          // 1/sqrt(64)
    for (int kt = 0; kt < Ntok; kt += 64) {
        __syncthreads();
        for (int i = threadIdx.x; i < 64 * HD; i += 128) {
            int r = i >> 6, c = i & 63;
            size_t g = ((size_t)bh * Ntok + kt + r) * HD + c;
            Ks[r][c] = Kb[g];
            Vs[r][c] = V[g];
        }
        __syncthreads();
        for (int j = 0; j < 64; ++j) {
            float s = 0.f;
#pragma unroll
            for (int e = 0; e < HD; ++e) s += q[e] * Ks[j][e];
            s *= scale;
            float mn   = fmaxf(mx, s);
            float corr = __expf(mx - mn);
            float p    = __expf(s - mn);
            l = l * corr + p;
#pragma unroll
            for (int e = 0; e < HD; ++e) acc[e] = acc[e] * corr + p * Vs[j][e];
            mx = mn;
        }
    }
    float inv = 1.f / l;
    _Float16* op = O + ((size_t)(b * Ntok + m) * HID) + h * HD;
#pragma unroll
    for (int e = 0; e < HD; ++e) op[e] = (_Float16)(acc[e] * inv);
}

// ============ f32 -> f16 weight convert with K padding ============
__global__ __launch_bounds__(256)
void dit_cvt(const float* __restrict__ src, _Float16* __restrict__ dst,
             int rows, int srcK, int dstK)
{
    size_t idx = (size_t)blockIdx.x * blockDim.x + threadIdx.x;
    if (idx >= (size_t)rows * dstK) return;
    int r = (int)(idx / dstK), k = (int)(idx - (size_t)r * dstK);
    dst[idx] = (k < srcK) ? (_Float16)src[(size_t)r * srcK + k] : (_Float16)0.f;
}

// ============ SwiGLU: g = silu(x1)*x2, padded to FFNP, fp16 ============
__global__ __launch_bounds__(256)
void dit_swiglu(const float* __restrict__ x12, _Float16* __restrict__ g)
{
    size_t idx = (size_t)blockIdx.x * blockDim.x + threadIdx.x;
    if (idx >= (size_t)Mrows * FFNP) return;
    int m = (int)(idx / FFNP), j = (int)(idx - (size_t)m * FFNP);
    float v = 0.f;
    if (j < FFN) {
        float a  = x12[(size_t)m * FFN2 + j];
        float b2 = x12[(size_t)m * FFN2 + FFN + j];
        v = a / (1.f + __expf(-a)) * b2;
    }
    g[idx] = (_Float16)v;
}

// ============ unpatchify -> out[B,C,H,W] ============
__global__ __launch_bounds__(256)
void dit_unpatch(const float* __restrict__ y, float* __restrict__ out)
{
    int idx = blockIdx.x * blockDim.x + threadIdx.x;
    if (idx >= Bz * Cc * Hh * Ww) return;
    int ww = idx & 63, t1 = idx >> 6, hh = t1 & 255, t2 = t1 >> 8, c = t2 & 1, b = t2 >> 1;
    int hp = hh >> 2, p = hh & 3, wp = ww >> 2, q = ww & 3;
    int n = hp * WP + wp;
    int o = (p * 4 + q) * 2 + c;
    out[idx] = y[((size_t)(b * Ntok + n)) * OUTC + o];
}

// =================== host orchestration ===================
static inline int ceil_div(long a, long b) { return (int)((a + b - 1) / b); }

extern "C" void kernel_launch(void* const* d_in, const int* in_sizes, int n_in,
                              void* d_out, int out_size, void* d_ws, size_t ws_size,
                              hipStream_t stream)
{
    const float* x       = (const float*)d_in[0];
    const float* t       = (const float*)d_in[1];
    const float* patch_w = (const float*)d_in[2];
    const float* patch_b = (const float*)d_in[3];
    const float* t_w     = (const float*)d_in[4];
    const float* t_b     = (const float*)d_in[5];
    const float* norm1_w = (const float*)d_in[6];
    const float* qkv_w   = (const float*)d_in[7];
    const float* qkv_b   = (const float*)d_in[8];
    const float* qn_w    = (const float*)d_in[9];
    const float* kn_w    = (const float*)d_in[10];
    const float* proj_w  = (const float*)d_in[11];
    const float* proj_b  = (const float*)d_in[12];
    const float* norm2_w = (const float*)d_in[13];
    const float* w12_w   = (const float*)d_in[14];
    const float* w12_b   = (const float*)d_in[15];
    const float* w3_w    = (const float*)d_in[16];
    const float* w3_b    = (const float*)d_in[17];
    const float* ada_w   = (const float*)d_in[18];
    const float* ada_b   = (const float*)d_in[19];
    const float* fnorm_w = (const float*)d_in[20];
    const float* flin_w  = (const float*)d_in[21];
    const float* flin_b  = (const float*)d_in[22];
    const float* fada_w  = (const float*)d_in[23];
    const float* fada_b  = (const float*)d_in[24];

    // workspace carve-up (256B aligned)
    char* base = (char*)d_ws;
    size_t off = 0;
    auto alloc = [&](size_t bytes) -> char* {
        char* p = base + off;
        off = (off + bytes + 255) & ~(size_t)255;
        return p;
    };
    float*     csP  = (float*)alloc((size_t)Bz * HID * 4);
    float*     adaP = (float*)alloc((size_t)Bz * 6 * HID * 4);          // also fada (8x2048)
    float*     hP   = (float*)alloc((size_t)Mrows * HID * 4);
    float*     bigP = (float*)alloc((size_t)Mrows * FFN2 * 4);          // qkv-out Y / x12
    _Float16*  ahP  = (_Float16*)alloc((size_t)Mrows * FFNP * 2);       // fp16 A: hn / O / g
    _Float16*  whP  = (_Float16*)alloc((size_t)FFN2 * HID * 2);         // fp16 weights
    float*     qP   = (float*)alloc((size_t)Bz * NH * Ntok * HD * 4);
    float*     kP   = (float*)alloc((size_t)Bz * NH * Ntok * HD * 4);
    float*     vP   = (float*)alloc((size_t)Bz * NH * Ntok * HD * 4);
    float*     yP   = (float*)alloc((size_t)Mrows * OUTC * 4);
    (void)ws_size;

    dim3 blk256(256);

    // timestep -> cs
    dit_tstep<<<ceil_div(Bz * HID, 256), blk256, 0, stream>>>(t, t_w, t_b, csP);
    // patchify + pos -> h
    dit_patch<<<ceil_div((long)Mrows * HID, 256), blk256, 0, stream>>>(x, patch_w, patch_b, hP);

    for (int d = 0; d < DEPTH; ++d) {
        // adaLN params [8, 6144]
        dit_ada<<<ceil_div(Bz * 6 * HID, 256), blk256, 0, stream>>>(
            csP, ada_w + (size_t)d * 6 * HID * HID, ada_b + (size_t)d * 6 * HID, adaP, 6 * HID);

        // hn1 = modulate(rmsnorm(h, norm1_w), sm, scm)  -> fp16
        dit_rmsmod<<<Mrows, blk256, 0, stream>>>(
            hP, norm1_w + (size_t)d * HID, adaP /*sm*/, adaP + HID /*scm*/, 6 * HID, ahP);

        // qkv GEMM: [8192,1024] x [3072,1024]^T
        dit_cvt<<<ceil_div((long)3 * HID * HID, 256), blk256, 0, stream>>>(
            qkv_w + (size_t)d * 3 * HID * HID, whP, 3 * HID, HID, HID);
        {
            dim3 g(ceil_div(3 * HID, 64), Mrows / 64);
            dit_gemm_wmma<<<g, dim3(128), 0, stream>>>(
                ahP, whP, qkv_b + (size_t)d * 3 * HID,
                nullptr, nullptr, 0, bigP, Mrows, 3 * HID, HID);
        }

        // split + q/k norm + rope
        dit_qkvpost<<<ceil_div(Bz * Ntok * NH, 256), blk256, 0, stream>>>(
            bigP, qn_w + (size_t)d * HD, kn_w + (size_t)d * HD, qP, kP, vP);

        // attention -> O fp16 (reuses ahP)
        dit_attn<<<dim3(Ntok / 128, Bz * NH), dim3(128), 0, stream>>>(qP, kP, vP, ahP);

        // proj GEMM + gated residual: h += gm * (O @ proj^T + b)
        dit_cvt<<<ceil_div((long)HID * HID, 256), blk256, 0, stream>>>(
            proj_w + (size_t)d * HID * HID, whP, HID, HID, HID);
        {
            dim3 g(ceil_div(HID, 64), Mrows / 64);
            dit_gemm_wmma<<<g, dim3(128), 0, stream>>>(
                ahP, whP, proj_b + (size_t)d * HID,
                hP, adaP + 2 * HID /*gm*/, 6 * HID, hP, Mrows, HID, HID);
        }

        // hn2 = modulate(rmsnorm(h, norm2_w), sp, scp) -> fp16
        dit_rmsmod<<<Mrows, blk256, 0, stream>>>(
            hP, norm2_w + (size_t)d * HID, adaP + 3 * HID /*sp*/, adaP + 4 * HID /*scp*/,
            6 * HID, ahP);

        // w12 GEMM: [8192,1024] x [5460,1024]^T -> x12
        dit_cvt<<<ceil_div((long)FFN2 * HID, 256), blk256, 0, stream>>>(
            w12_w + (size_t)d * FFN2 * HID, whP, FFN2, HID, HID);
        {
            dim3 g(ceil_div(FFN2, 64), Mrows / 64);
            dit_gemm_wmma<<<g, dim3(128), 0, stream>>>(
                ahP, whP, w12_b + (size_t)d * FFN2,
                nullptr, nullptr, 0, bigP, Mrows, FFN2, HID);
        }

        // SwiGLU -> g fp16 padded [8192, 2752]
        dit_swiglu<<<ceil_div((long)Mrows * FFNP, 256), blk256, 0, stream>>>(bigP, ahP);

        // w3 GEMM (padded K) + gated residual: h += gp * (g @ w3^T + b)
        dit_cvt<<<ceil_div((long)HID * FFNP, 256), blk256, 0, stream>>>(
            w3_w + (size_t)d * HID * FFN, whP, HID, FFN, FFNP);
        {
            dim3 g(ceil_div(HID, 64), Mrows / 64);
            dit_gemm_wmma<<<g, dim3(128), 0, stream>>>(
                ahP, whP, w3_b + (size_t)d * HID,
                hP, adaP + 5 * HID /*gp*/, 6 * HID, hP, Mrows, HID, FFNP);
        }
    }

    // final adaLN [8, 2048]
    dit_ada<<<ceil_div(Bz * 2 * HID, 256), blk256, 0, stream>>>(
        csP, fada_w, fada_b, adaP, 2 * HID);
    // final modulate -> fp16
    dit_rmsmod<<<Mrows, blk256, 0, stream>>>(
        hP, fnorm_w, adaP /*fsh*/, adaP + HID /*fsc*/, 2 * HID, ahP);
    // final GEMM: [8192,1024] x [32,1024]^T
    dit_cvt<<<ceil_div((long)OUTC * HID, 256), blk256, 0, stream>>>(
        flin_w, whP, OUTC, HID, HID);
    {
        dim3 g(ceil_div(OUTC, 64), Mrows / 64);
        dit_gemm_wmma<<<g, dim3(128), 0, stream>>>(
            ahP, whP, flin_b, nullptr, nullptr, 0, yP, Mrows, OUTC, HID);
    }
    // unpatchify -> out
    dit_unpatch<<<ceil_div(Bz * Cc * Hh * Ww, 256), blk256, 0, stream>>>(yP, (float*)d_out);
}
